// Weight_Sample_Module_4037269258369
// MI455X (gfx1250) — matrix operand order repaired and checked by reading, not compiled, not run on hardware
//
#include <hip/hip_runtime.h>
#include <hip/hip_bf16.h>

// ---- static problem configuration (matches reference) ----
constexpr int BC    = 4;      // clouds
constexpr int NPER  = 4096;   // points per cloud
constexpr int MPER  = 1024;   // sampled centroids per cloud
constexpr int MTOT  = BC * MPER;       // 4096 centroids total
constexpr int KNB   = 64;     // max neighbors
constexpr int CF    = 64;     // feature dim
constexpr int CIN   = 67;     // CF + 3
constexpr int CINP  = 96;     // padded to multiple of 32 (3 K-steps)
constexpr int H1D   = 128;
constexpr int H2D   = 256;
constexpr float R2F = 0.04f;  // radius^2

typedef __attribute__((ext_vector_type(16))) __bf16 v16bf;
typedef __attribute__((ext_vector_type(8)))  __bf16 v8bf;
typedef __attribute__((ext_vector_type(8)))  float  v8f;

// ---------------------------------------------------------------------------
// Kernel 0: convert weights to bf16, TRANSPOSED to N-major so that a WMMA
// B-tile (fixed N per lane, 16 consecutive K) is one contiguous 32-byte load.
//   W1pT[n][k] : [H1D][CINP], k >= CIN zero-padded
//   W2pT[n][k] : [H2D][H1D]
// ---------------------------------------------------------------------------
__global__ void convert_weights_kernel(const float* __restrict__ W1,
                                       const float* __restrict__ W2,
                                       __bf16* __restrict__ W1pT,
                                       __bf16* __restrict__ W2pT) {
  int t = blockIdx.x * blockDim.x + threadIdx.x;
  if (t < H1D * CINP) {
    int n = t / CINP, k = t % CINP;
    W1pT[t] = (k < CIN) ? (__bf16)W1[k * H1D + n] : (__bf16)0.0f;
  }
  if (t < H2D * H1D) {
    int n = t / H1D, k = t % H1D;
    W2pT[t] = (__bf16)W2[k * H2D + n];
  }
}

// ---------------------------------------------------------------------------
// Kernel 1: farthest point sampling, one block per cloud.
// Also emits pos_out / batch_out / weight_out for the sampled points.
// ---------------------------------------------------------------------------
__global__ void __launch_bounds__(256)
fps_kernel(const float* __restrict__ pos, const long long* __restrict__ batch,
           const float* __restrict__ weight, int* __restrict__ idx,
           float* __restrict__ out_pos, float* __restrict__ out_batch,
           float* __restrict__ out_weight) {
  __shared__ float px[NPER], py[NPER], pz[NPER];   // 48 KB
  __shared__ float red_v[256];
  __shared__ int   red_i[256];

  const int b   = blockIdx.x;
  const int tid = threadIdx.x;
  const float* p = pos + (size_t)b * NPER * 3;

  float mind[NPER / 256];                 // 16 per thread, strided i = tid + 256*j
  for (int j = 0; j < NPER / 256; ++j) {
    int i = tid + 256 * j;
    px[i] = p[i * 3 + 0];
    py[i] = p[i * 3 + 1];
    pz[i] = p[i * 3 + 2];
    mind[j] = __builtin_inff();
  }
  __syncthreads();

  int last = 0;  // deterministic start at local index 0 (matches reference)
  for (int s = 0; s < MPER; ++s) {
    if (tid == 0) {
      int g = b * NPER + last;
      int m = b * MPER + s;
      idx[m]             = g;
      out_pos[m * 3 + 0] = px[last];
      out_pos[m * 3 + 1] = py[last];
      out_pos[m * 3 + 2] = pz[last];
      out_batch[m]       = (float)batch[g];
      out_weight[m]      = weight[g];
    }
    const float lx = px[last], ly = py[last], lz = pz[last];
    float bestv = -1.0f; int besti = 0;
    for (int j = 0; j < NPER / 256; ++j) {
      int i = tid + 256 * j;
      float dx = px[i] - lx, dy = py[i] - ly, dz = pz[i] - lz;
      float d2 = dx * dx + dy * dy + dz * dz;
      float m  = fminf(mind[j], d2);
      mind[j]  = m;
      if (m > bestv) { bestv = m; besti = i; }
    }
    red_v[tid] = bestv;
    red_i[tid] = besti;
    __syncthreads();
    for (int off = 128; off > 0; off >>= 1) {
      if (tid < off) {
        if (red_v[tid + off] > red_v[tid]) {
          red_v[tid] = red_v[tid + off];
          red_i[tid] = red_i[tid + off];
        }
      }
      __syncthreads();
    }
    last = red_i[0];
    __syncthreads();   // protect red_* WAR for next iteration
  }
}

// ---------------------------------------------------------------------------
// Kernel 2: ball query, one wave32 per centroid; deterministic ordered
// append via ballot/popcount (no atomics). Unfilled slots get -1.
// ---------------------------------------------------------------------------
__global__ void __launch_bounds__(256)
ballquery_kernel(const float* __restrict__ pos, const int* __restrict__ idx,
                 int* __restrict__ nbr) {
  const int wv   = threadIdx.x / 32;
  const int lane = threadIdx.x % 32;
  const int cid  = blockIdx.x * 8 + wv;
  const int g    = idx[cid];
  const int b    = g / NPER;
  const float* p = pos + (size_t)b * NPER * 3;
  const float qx = pos[g * 3 + 0], qy = pos[g * 3 + 1], qz = pos[g * 3 + 2];

  int cnt = 0;
  for (int base = 0; base < NPER; base += 32) {
    int i = base + lane;
    float dx = p[i * 3 + 0] - qx;
    float dy = p[i * 3 + 1] - qy;
    float dz = p[i * 3 + 2] - qz;
    bool hit = (dx * dx + dy * dy + dz * dz) <= R2F;
    unsigned long long m = __ballot(hit);
    int before = __popcll(m & ((1ull << lane) - 1ull));
    if (hit) {
      int slot = cnt + before;
      if (slot < KNB) nbr[(size_t)cid * KNB + slot] = b * NPER + i;
    }
    cnt += __popcll(m);
    if (cnt >= KNB) break;      // uniform across wave
  }
  int c = cnt < KNB ? cnt : KNB;
  for (int s = c + lane; s < KNB; s += 32) nbr[(size_t)cid * KNB + s] = -1;
}

// ---------------------------------------------------------------------------
// helpers for WMMA fragment loads
// ---------------------------------------------------------------------------
// A-tile fragment: element e of lane L holds
//   K = ks*32 + 16*half + e,  M = 16*mt + L%16
// -> one contiguous 32-byte chunk per lane (matches B fragment K order).
__device__ inline v16bf load_a_frag(const __bf16* row_k0, int half) {
  return *(const v16bf*)(row_k0 + 16 * half);
}
// B-tile fragment from N-major weights: element e holds K = ks*32 + 16*half + e,
// N = L%16 -> 16 consecutive bf16 = one 32-byte contiguous load.
__device__ inline v16bf load_b_frag(const __bf16* wT_row_n, int ks, int half) {
  return *(const v16bf*)(wT_row_n + ks * 32 + 16 * half);
}

// ---------------------------------------------------------------------------
// Kernel 3: PointNetConv MLP + masked max aggregation.
// One block (128 threads = 4 waves) per centroid, bf16 WMMA, f32 accumulate.
// ---------------------------------------------------------------------------
__global__ void __launch_bounds__(128)
mlp_kernel(const float* __restrict__ x, const float* __restrict__ pos,
           const int* __restrict__ idx, const int* __restrict__ nbr,
           const __bf16* __restrict__ W1pT, const __bf16* __restrict__ W2pT,
           const float* __restrict__ b1, const float* __restrict__ b2,
           float* __restrict__ x_out) {
  __shared__ __bf16 msg[KNB][CINP];   // 64 x 96  = 12 KB
  __shared__ __bf16 h1[KNB][H1D];     // 64 x 128 = 16 KB
  __shared__ int      s_nbr[KNB];
  __shared__ unsigned s_vmask[2];

  const int tid = threadIdx.x;
  const int cid = blockIdx.x;
  const int g   = idx[cid];
  const float qx = pos[g * 3 + 0], qy = pos[g * 3 + 1], qz = pos[g * 3 + 2];

  if (tid < 2) s_vmask[tid] = 0u;
  __syncthreads();
  if (tid < KNB) {
    int nb = nbr[(size_t)cid * KNB + tid];
    bool valid = nb >= 0;
    if (!valid) nb = g;                 // safe gather target for invalid slots
    s_nbr[tid] = nb;
    if (valid) atomicOr(&s_vmask[tid / 32], 1u << (tid % 32));
  }
  __syncthreads();

  // stage msg = concat(x_j, pos_j - qpos), padded to 96 cols, as bf16
  for (int e = tid; e < KNB * CINP; e += 128) {
    int r = e / CINP, c = e % CINP;
    int nb = s_nbr[r];
    float v;
    if (c < CF)       v = x[(size_t)nb * CF + c];
    else if (c == 64) v = pos[nb * 3 + 0] - qx;
    else if (c == 65) v = pos[nb * 3 + 1] - qy;
    else if (c == 66) v = pos[nb * 3 + 2] - qz;
    else              v = 0.0f;
    msg[r][c] = (__bf16)v;
  }
  __syncthreads();

  const int wv   = tid / 32;
  const int lane = tid % 32;
  const int half = lane / 16;      // ISA layout: lanes 16-31 are second halves
  const int lm   = lane % 16;

  // ---- layer 1: H1 = ReLU(msg @ W1 + b1); wave wv owns 32 output columns ----
  for (int nt = 2 * wv; nt < 2 * wv + 2; ++nt) {
    const int n    = nt * 16 + lm;
    const float bs = b1[n];
    const __bf16* wrow = W1pT + (size_t)n * CINP;
    for (int mt = 0; mt < 4; ++mt) {
      v8f acc = {};
      const int m = mt * 16 + lm;
      for (int ks = 0; ks < 3; ++ks) {
        v16bf a  = load_a_frag(&msg[m][ks * 32], half);
        v16bf bm = load_b_frag(wrow, ks, half);
        acc = __builtin_amdgcn_wmma_f32_16x16x32_bf16(
            false, a, false, bm, (short)0, acc, false, false);
      }
#pragma unroll
      for (int r = 0; r < 8; ++r) {
        float v = acc[r] + bs;
        v = v > 0.0f ? v : 0.0f;
        h1[mt * 16 + r + 8 * half][n] = (__bf16)v;
      }
    }
  }
  __syncthreads();

  // ---- layer 2 + masked max over the 64 neighbor rows (fused epilogue) ----
  const unsigned vm0 = s_vmask[0], vm1 = s_vmask[1];
  for (int nt = 4 * wv; nt < 4 * wv + 4; ++nt) {
    const int n    = nt * 16 + lm;
    const float bs = b2[n];
    const __bf16* wrow = W2pT + (size_t)n * H1D;
    float lmax = -__builtin_inff();
    for (int mt = 0; mt < 4; ++mt) {
      v8f acc = {};
      const int m = mt * 16 + lm;
      for (int ks = 0; ks < 4; ++ks) {
        v16bf a  = load_a_frag(&h1[m][ks * 32], half);
        v16bf bm = load_b_frag(wrow, ks, half);
        acc = __builtin_amdgcn_wmma_f32_16x16x32_bf16(
            false, a, false, bm, (short)0, acc, false, false);
      }
#pragma unroll
      for (int r = 0; r < 8; ++r) {
        int mrow = mt * 16 + r + 8 * half;
        unsigned bit = (mrow < 32) ? (vm0 >> mrow) & 1u : (vm1 >> (mrow - 32)) & 1u;
        float v = acc[r] + bs;
        v = v > 0.0f ? v : 0.0f;
        if (bit) lmax = fmaxf(lmax, v);
      }
    }
    // merge the two M-halves (lanes L and L+16 hold the same N)
    float o = __shfl_xor(lmax, 16, 32);
    lmax = fmaxf(lmax, o);
    if (half == 0) x_out[(size_t)cid * H2D + n] = lmax;
  }
}

// ---------------------------------------------------------------------------
// launch
// ---------------------------------------------------------------------------
extern "C" void kernel_launch(void* const* d_in, const int* in_sizes, int n_in,
                              void* d_out, int out_size, void* d_ws, size_t ws_size,
                              hipStream_t stream) {
  const float*     x      = (const float*)d_in[0];
  const float*     pos    = (const float*)d_in[1];
  const long long* batch  = (const long long*)d_in[2];
  const float*     weight = (const float*)d_in[3];
  const float*     W1     = (const float*)d_in[4];
  const float*     b1     = (const float*)d_in[5];
  const float*     W2     = (const float*)d_in[6];
  const float*     b2     = (const float*)d_in[7];

  // workspace layout (offsets are 32-byte multiples for vector loads)
  char* ws = (char*)d_ws;
  int*    idx  = (int*)ws;                                    // 4096 * 4
  int*    nbr  = (int*)(ws + MTOT * sizeof(int));             // 4096*64*4
  __bf16* W1pT = (__bf16*)(ws + MTOT * sizeof(int) + (size_t)MTOT * KNB * sizeof(int));
  __bf16* W2pT = W1pT + H1D * CINP;

  // output layout: x_out | pos_out | batch_out | weight_out (all f32)
  float* out        = (float*)d_out;
  float* x_out      = out;
  float* pos_out    = x_out + (size_t)MTOT * H2D;
  float* batch_out  = pos_out + (size_t)MTOT * 3;
  float* weight_out = batch_out + MTOT;

  convert_weights_kernel<<<(H2D * H1D + 255) / 256, 256, 0, stream>>>(W1, W2, W1pT, W2pT);
  fps_kernel<<<BC, 256, 0, stream>>>(pos, batch, weight, idx, pos_out, batch_out, weight_out);
  ballquery_kernel<<<MTOT / 8, 256, 0, stream>>>(pos, idx, nbr);
  mlp_kernel<<<MTOT, 128, 0, stream>>>(x, pos, idx, nbr, W1pT, W2pT, b1, b2, x_out);
}
